// InteractionMHA_66752381714915
// MI455X (gfx1250) — matrix operand order repaired and checked by reading, not compiled, not run on hardware
//
#include <hip/hip_runtime.h>
#include <math.h>

typedef __attribute__((ext_vector_type(16))) _Float16 v16h;
typedef __attribute__((ext_vector_type(8)))  _Float16 v8h;
typedef __attribute__((ext_vector_type(8)))  float    v8f;

static __device__ __forceinline__ v16h join8(v8h a, v8h b) {
  return __builtin_shufflevector(a, b, 0,1,2,3,4,5,6,7,8,9,10,11,12,13,14,15);
}

static __device__ __forceinline__ float gelu_exact(float x) {
  return 0.5f * x * (1.0f + erff(x * 0.70710678118654752f));
}

// ---------------------------------------------------------------------------
// Offsets: exclusive prefix sum of num_atoms (B small: single thread is fine)
// ---------------------------------------------------------------------------
__global__ void compute_offsets_kernel(const int* __restrict__ num_atoms,
                                       int* __restrict__ offsets, int B) {
  if (blockIdx.x == 0 && threadIdx.x == 0) {
    int acc = 0;
    for (int i = 0; i < B; ++i) { offsets[i] = acc; acc += num_atoms[i]; }
    offsets[B] = acc;
  }
}

// ---------------------------------------------------------------------------
// GEMM: Y[M,256] = act(X[M,256] @ W[256,256]^T + bias)
// InT = float (converted to f16 on LDS stage) or _Float16.
// Block: 256 threads (8 waves), tile 64(M) x 64(N), K-loop in chunks of 32.
// Each wave: 16x32 output = two v_wmma_f32_16x16x32_f16 per k-step.
// ---------------------------------------------------------------------------
template <typename InT, typename OutT, int ACT>
__global__ __launch_bounds__(256) void gemm256_kernel(
    const InT* __restrict__ X, const float* __restrict__ W,
    const float* __restrict__ bias, OutT* __restrict__ Y, int M) {
  __shared__ __align__(16) _Float16 sX[64 * 40];
  __shared__ __align__(16) _Float16 sW[64 * 40];

  const int tid   = threadIdx.x;
  const int lane  = tid & 31;
  const int wave  = tid >> 5;
  const int l16   = lane & 15;
  const int halfg = lane >> 4;

  const int m0 = blockIdx.x * 64;
  const int n0 = blockIdx.y * 64;

  const int mSub = (wave & 3) * 16;
  const int nSub = (wave >> 2) * 32;

  v8f acc0 = {}, acc1 = {};

  const int ldr = tid >> 2;          // 0..63 (tile row)
  const int ldc = (tid & 3) * 8;     // 0,8,16,24 (k-col base)

  for (int kt = 0; kt < 8; ++kt) {
    const int k0 = kt * 32;
    // ---- stage X tile (64x32) ----
    {
      v8h h = {};
      const int gm = m0 + ldr;
      if (gm < M) {
        if constexpr (sizeof(InT) == 4) {
          const float4 a = *(const float4*)(X + (size_t)gm * 256 + k0 + ldc);
          const float4 b = *(const float4*)(X + (size_t)gm * 256 + k0 + ldc + 4);
          h[0]=(_Float16)a.x; h[1]=(_Float16)a.y; h[2]=(_Float16)a.z; h[3]=(_Float16)a.w;
          h[4]=(_Float16)b.x; h[5]=(_Float16)b.y; h[6]=(_Float16)b.z; h[7]=(_Float16)b.w;
        } else {
          h = *(const v8h*)(X + (size_t)gm * 256 + k0 + ldc);
        }
      }
      *(v8h*)(sX + ldr * 40 + ldc) = h;
    }
    // ---- stage W tile (64x32), f32 -> f16 ----
    {
      const float4 a = *(const float4*)(W + (size_t)(n0 + ldr) * 256 + k0 + ldc);
      const float4 b = *(const float4*)(W + (size_t)(n0 + ldr) * 256 + k0 + ldc + 4);
      v8h h;
      h[0]=(_Float16)a.x; h[1]=(_Float16)a.y; h[2]=(_Float16)a.z; h[3]=(_Float16)a.w;
      h[4]=(_Float16)b.x; h[5]=(_Float16)b.y; h[6]=(_Float16)b.z; h[7]=(_Float16)b.w;
      *(v8h*)(sW + ldr * 40 + ldc) = h;
    }
    __syncthreads();

    // ---- WMMA ----
    const _Float16* xr = sX + (mSub + l16) * 40;
    v16h af = join8(*(const v8h*)(xr + halfg * 8),
                    *(const v8h*)(xr + 16 + halfg * 8));
    const _Float16* wr0 = sW + (nSub + l16) * 40 + halfg * 8;
    v16h bf0 = join8(*(const v8h*)wr0, *(const v8h*)(wr0 + 16));
    const _Float16* wr1 = sW + (nSub + 16 + l16) * 40 + halfg * 8;
    v16h bf1 = join8(*(const v8h*)wr1, *(const v8h*)(wr1 + 16));

    acc0 = __builtin_amdgcn_wmma_f32_16x16x32_f16(false, af, false, bf0,
                                                  (short)0, acc0, false, false);
    acc1 = __builtin_amdgcn_wmma_f32_16x16x32_f16(false, af, false, bf1,
                                                  (short)0, acc1, false, false);
    __syncthreads();
  }

  // ---- epilogue: bias + activation + store ----
#pragma unroll
  for (int j = 0; j < 8; ++j) {
    const int m = m0 + mSub + halfg * 8 + j;
    if (m >= M) continue;
    const int na = n0 + nSub + l16;
    const int nb = na + 16;
    float ya = acc0[j] + bias[na];
    float yb = acc1[j] + bias[nb];
    if constexpr (ACT == 1) { ya = gelu_exact(ya); yb = gelu_exact(yb); }
    Y[(size_t)m * 256 + na] = (OutT)ya;
    Y[(size_t)m * 256 + nb] = (OutT)yb;
  }
}

// ---------------------------------------------------------------------------
// Attention: one workgroup per graph. N <= 128, D = 256.
// K staged row-major in LDS; Q strip kept in registers as WMMA A-fragments;
// scores + softmax fully in registers (half-wave shuffle reductions);
// A -> f16 in LDS; V staged transposed in LDS; O = A @ V via WMMA.
// ---------------------------------------------------------------------------
__global__ __launch_bounds__(256) void attn_kernel(
    const _Float16* __restrict__ Q, const _Float16* __restrict__ K,
    const _Float16* __restrict__ V, const int* __restrict__ offsets,
    const int* __restrict__ num_atoms, _Float16* __restrict__ Xout) {
  // region0: K rows [128][272] OR V^T [256][136]  (both 34816 halves)
  // region1: A f16 [128][136]
  __shared__ __align__(16) _Float16 smem[256 * 136 + 128 * 136];
  _Float16* sKV = smem;
  _Float16* sA  = smem + 256 * 136;

  const int b    = blockIdx.x;
  const int base = offsets[b];
  const int N    = num_atoms[b];
  const int tid  = threadIdx.x;
  const int lane = tid & 31;
  const int wave = tid >> 5;
  const int l16  = lane & 15;
  const int halfg = lane >> 4;

  // ---- stage K (row-major, stride 272 halves), zero-pad rows >= N ----
  for (int c = tid; c < 128 * 32; c += 256) {
    const int r   = c >> 5;
    const int col = (c & 31) * 8;
    v8h val = {};
    if (r < N) val = *(const v8h*)(K + (size_t)(base + r) * 256 + col);
    *(v8h*)(sKV + r * 272 + col) = val;
  }

  // ---- Q strip for this wave: rows wave*16 .. +15, full K=256 in 8 frags ----
  const int mrow = wave * 16 + l16;
  const bool qvalid = (mrow < N);
  v16h qf[8];
#pragma unroll
  for (int kt = 0; kt < 8; ++kt) {
    v8h lo = {}, hi = {};
    if (qvalid) {
      const _Float16* qr = Q + (size_t)(base + mrow) * 256 + kt * 32;
      lo = *(const v8h*)(qr + halfg * 8);
      hi = *(const v8h*)(qr + 16 + halfg * 8);
    }
    qf[kt] = join8(lo, hi);
  }
  __syncthreads();

  // ---- S = Q @ K^T  (8 key tiles of 16, each over 8 k-steps) ----
  v8f acc[8];
#pragma unroll
  for (int nt = 0; nt < 8; ++nt) {
    v8f c = {};
#pragma unroll
    for (int kt = 0; kt < 8; ++kt) {
      const _Float16* kr = sKV + (nt * 16 + l16) * 272 + kt * 32 + halfg * 8;
      v16h bf = join8(*(const v8h*)kr, *(const v8h*)(kr + 16));
      c = __builtin_amdgcn_wmma_f32_16x16x32_f16(false, qf[kt], false, bf,
                                                 (short)0, c, false, false);
    }
    acc[nt] = c;
  }

  // ---- scale + key-padding mask ----
  const float scale = 0.0625f;  // 1/sqrt(256)
#pragma unroll
  for (int nt = 0; nt < 8; ++nt) {
    const bool kvalid = (nt * 16 + l16) < N;
#pragma unroll
    for (int j = 0; j < 8; ++j)
      acc[nt][j] = kvalid ? acc[nt][j] * scale : -1e30f;
  }

  // ---- softmax per row: row lives in vgpr j across the 16 lanes of a half ----
#pragma unroll
  for (int j = 0; j < 8; ++j) {
    float mx = -1e30f;
#pragma unroll
    for (int nt = 0; nt < 8; ++nt) mx = fmaxf(mx, acc[nt][j]);
    for (int s = 1; s < 16; s <<= 1) mx = fmaxf(mx, __shfl_xor(mx, s, 32));
    float sum = 0.f;
#pragma unroll
    for (int nt = 0; nt < 8; ++nt) {
      const float e = __expf(acc[nt][j] - mx);
      acc[nt][j] = e; sum += e;
    }
    for (int s = 1; s < 16; s <<= 1) sum += __shfl_xor(sum, s, 32);
    const float inv = 1.0f / sum;
#pragma unroll
    for (int nt = 0; nt < 8; ++nt) acc[nt][j] *= inv;
  }

  // ---- write A (f16) to LDS, stride 136 halves ----
#pragma unroll
  for (int nt = 0; nt < 8; ++nt)
#pragma unroll
    for (int j = 0; j < 8; ++j)
      sA[(wave * 16 + halfg * 8 + j) * 136 + nt * 16 + l16] =
          (_Float16)acc[nt][j];

  __syncthreads();

  // ---- stage V transposed: sVt[col][row], stride 136 halves ----
  for (int c = tid; c < 128 * 32; c += 256) {
    const int r    = c >> 5;
    const int colb = (c & 31) * 8;
    v8h val = {};
    if (r < N) val = *(const v8h*)(V + (size_t)(base + r) * 256 + colb);
#pragma unroll
    for (int i = 0; i < 8; ++i) sKV[(colb + i) * 136 + r] = val[i];
  }
  __syncthreads();

  // ---- A-fragments for this wave's 16 rows (K = 128 keys, 4 frags) ----
  v16h afr[4];
#pragma unroll
  for (int kt = 0; kt < 4; ++kt) {
    const _Float16* ar = sA + (wave * 16 + l16) * 136 + kt * 32 + halfg * 8;
    afr[kt] = join8(*(const v8h*)ar, *(const v8h*)(ar + 16));
  }

  // ---- O = A @ V : 4 column chunks of 64 ----
  for (int nc = 0; nc < 4; ++nc) {
    v8f oc[4] = {v8f{}, v8f{}, v8f{}, v8f{}};
#pragma unroll
    for (int kt = 0; kt < 4; ++kt) {
#pragma unroll
      for (int t = 0; t < 4; ++t) {
        const _Float16* vr =
            sKV + (nc * 64 + t * 16 + l16) * 136 + kt * 32 + halfg * 8;
        v16h bf = join8(*(const v8h*)vr, *(const v8h*)(vr + 16));
        oc[t] = __builtin_amdgcn_wmma_f32_16x16x32_f16(false, afr[kt], false, bf,
                                                       (short)0, oc[t], false, false);
      }
    }
#pragma unroll
    for (int t = 0; t < 4; ++t) {
#pragma unroll
      for (int j = 0; j < 8; ++j) {
        const int m = wave * 16 + halfg * 8 + j;
        if (m < N)
          Xout[(size_t)(base + m) * 256 + nc * 64 + t * 16 + l16] =
              (_Float16)oc[t][j];
      }
    }
  }
}

// ---------------------------------------------------------------------------
extern "C" void kernel_launch(void* const* d_in, const int* in_sizes, int n_in,
                              void* d_out, int out_size, void* d_ws,
                              size_t ws_size, hipStream_t stream) {
  const float* rep2d = (const float*)d_in[0];
  const float* rep3d = (const float*)d_in[1];
  const int* num_atoms = (const int*)d_in[2];
  const float* wq23 = (const float*)d_in[3];  const float* bq23 = (const float*)d_in[4];
  const float* wk23 = (const float*)d_in[5];  const float* bk23 = (const float*)d_in[6];
  const float* wv23 = (const float*)d_in[7];  const float* bv23 = (const float*)d_in[8];
  const float* wq32 = (const float*)d_in[9];  const float* bq32 = (const float*)d_in[10];
  const float* wk32 = (const float*)d_in[11]; const float* bk32 = (const float*)d_in[12];
  const float* wv32 = (const float*)d_in[13]; const float* bv32 = (const float*)d_in[14];
  const float* wm23 = (const float*)d_in[15]; const float* bm23 = (const float*)d_in[16];
  const float* wm32 = (const float*)d_in[17]; const float* bm32 = (const float*)d_in[18];

  const int total = in_sizes[0] / 256;
  const int B = in_sizes[2];
  const size_t T = (size_t)total * 256;

  int* offsets = (int*)d_ws;
  _Float16* f = (_Float16*)((char*)d_ws + 4096);
  _Float16* q23 = f + 0 * T; _Float16* k23 = f + 1 * T; _Float16* v23 = f + 2 * T;
  _Float16* q32 = f + 3 * T; _Float16* k32 = f + 4 * T; _Float16* v32 = f + 5 * T;
  _Float16* x23 = f + 6 * T; _Float16* x32 = f + 7 * T;

  compute_offsets_kernel<<<1, 64, 0, stream>>>(num_atoms, offsets, B);

  dim3 gg((total + 63) / 64, 4), bb(256);
  // QKV projections (f16 out, no activation)
  gemm256_kernel<float, _Float16, 0><<<gg, bb, 0, stream>>>(rep2d, wq23, bq23, q23, total);
  gemm256_kernel<float, _Float16, 0><<<gg, bb, 0, stream>>>(rep3d, wk23, bk23, k23, total);
  gemm256_kernel<float, _Float16, 0><<<gg, bb, 0, stream>>>(rep3d, wv23, bv23, v23, total);
  gemm256_kernel<float, _Float16, 0><<<gg, bb, 0, stream>>>(rep3d, wq32, bq32, q32, total);
  gemm256_kernel<float, _Float16, 0><<<gg, bb, 0, stream>>>(rep2d, wk32, bk32, k32, total);
  gemm256_kernel<float, _Float16, 0><<<gg, bb, 0, stream>>>(rep2d, wv32, bv32, v32, total);

  // Cross attentions: (2d queries, 3d kv) and (3d queries, 2d kv)
  attn_kernel<<<B, 256, 0, stream>>>(q23, k23, v23, offsets, num_atoms, x23);
  attn_kernel<<<B, 256, 0, stream>>>(q32, k32, v32, offsets, num_atoms, x32);

  // Output linears + exact GELU (f32 out to d_out, concatenated)
  gemm256_kernel<_Float16, float, 1><<<gg, bb, 0, stream>>>(x23, wm23, bm23, (float*)d_out, total);
  gemm256_kernel<_Float16, float, 1><<<gg, bb, 0, stream>>>(x32, wm32, bm32, (float*)d_out + T, total);
}
